// MultiHeadAttentionParallel_249108103499
// MI455X (gfx1250) — matrix operand order repaired and checked by reading, not compiled, not run on hardware
//
#include <hip/hip_runtime.h>
#include <hip/hip_bf16.h>

typedef _Float16 h16 __attribute__((ext_vector_type(16)));
typedef _Float16 h8v __attribute__((ext_vector_type(8)));
typedef float    f8v __attribute__((ext_vector_type(8)));

// ---------------------------------------------------------------------------
// WMMA helpers: V_WMMA_F32_16X16X32_F16, fragments per CDNA5 ISA layouts.
// A (16x32, row-major source): lane l: m=l&15, sel=l>>4; halves 0..7 = K[sel*8..+7],
//                              halves 8..15 = K[16+sel*8..+7]  -> two b128 loads.
// B (32x16) from TRANSPOSED storage Bt[n][k]: lane l: n=l&15, sel=l>>4;
//                              halves 0..7 -> K = k0 + sel*8 + h, 8..15 -> K = k0 + 16 + sel*8 + (h-8) -> two b128 loads.
// D (16x16 f32): lane l: n=l&15; vgpr r -> m = r + 8*(l>>4).
// ---------------------------------------------------------------------------
__device__ __forceinline__ f8v wmma_f(h16 a, h16 b, f8v c) {
  return __builtin_amdgcn_wmma_f32_16x16x32_f16(false, a, false, b, (short)0, c,
                                                false, false);
}

__device__ __forceinline__ h16 frag_a(const _Float16* __restrict__ base, int ld,
                                      int row0, int k0) {
  const int l = threadIdx.x & 31;
  const int m = l & 15, sel = l >> 4;
  const _Float16* p = base + (size_t)(row0 + m) * ld + k0 + sel * 8;
  h8v lo = *reinterpret_cast<const h8v*>(p);
  h8v hi = *reinterpret_cast<const h8v*>(p + 16);
  return __builtin_shufflevector(lo, hi, 0, 1, 2, 3, 4, 5, 6, 7,
                                 8, 9, 10, 11, 12, 13, 14, 15);
}

__device__ __forceinline__ h16 frag_bt(const _Float16* __restrict__ baseT, int ld,
                                       int n0, int k0) {
  const int l = threadIdx.x & 31;
  const int n = l & 15, sel = l >> 4;
  const _Float16* p = baseT + (size_t)(n0 + n) * ld + k0 + sel * 8;
  h8v lo = *reinterpret_cast<const h8v*>(p);
  h8v hi = *reinterpret_cast<const h8v*>(p + 16);
  return __builtin_shufflevector(lo, hi, 0, 1, 2, 3, 4, 5, 6, 7,
                                 8, 9, 10, 11, 12, 13, 14, 15);
}

// ---------------------------------------------------------------------------
// Precision conversion / weight transpose (one-shot, tiny vs GEMM cost)
// ---------------------------------------------------------------------------
__global__ void k_f32_to_f16(const float* __restrict__ src,
                             _Float16* __restrict__ dst, int n) {
  int i = blockIdx.x * blockDim.x + threadIdx.x;
  if (i < n) dst[i] = (_Float16)src[i];
}

// src [R x C] row-major f32 -> dst [C x R] row-major f16 (i.e. transposed)
__global__ void k_transpose_f16(const float* __restrict__ src,
                                _Float16* __restrict__ dst, int R, int C) {
  int i = blockIdx.x * blockDim.x + threadIdx.x;
  if (i < R * C) {
    int r = i / C, c = i % C;
    dst[(size_t)c * R + r] = (_Float16)src[i];
  }
}

// ---------------------------------------------------------------------------
// GEMM1: qkv = xh[65536x384] @ Wqkv, scatter-store into
//   Qb [B,H,T,64], Kb [B,H,T,64], Vt [B,H,64,T]  (V pre-transposed for PV GEMM)
// grid (512, 18), block 256 (8 waves): 128x64 tile per WG, 16x64 per wave.
// ---------------------------------------------------------------------------
__global__ void __launch_bounds__(256) k_gemm_qkv(
    const _Float16* __restrict__ xh, const _Float16* __restrict__ WT,
    _Float16* __restrict__ Qb, _Float16* __restrict__ Kb,
    _Float16* __restrict__ Vt) {
  const int wave = threadIdx.x >> 5, l = threadIdx.x & 31;
  const int row0 = blockIdx.x * 128 + wave * 16;
  const int n0 = blockIdx.y * 64;
  f8v c[4] = {};
  for (int k0 = 0; k0 < 384; k0 += 32) {
    h16 a = frag_a(xh, 384, row0, k0);
#pragma unroll
    for (int t = 0; t < 4; ++t) {
      h16 b = frag_bt(WT, 384, n0 + t * 16, k0);
      c[t] = wmma_f(a, b, c[t]);
    }
  }
  const int n = l & 15, sel = l >> 4;
#pragma unroll
  for (int t = 0; t < 4; ++t) {
    const int col = n0 + t * 16 + n;
#pragma unroll
    for (int r = 0; r < 8; ++r) {
      const int row = row0 + r + 8 * sel;          // global token row
      const int bI = row >> 8, tok = row & 255;
      const _Float16 v = (_Float16)c[t][r];
      if (col < 384) {                              // Q
        const int hI = col >> 6, d = col & 63;
        Qb[(((size_t)bI * 6 + hI) * 256 + tok) * 64 + d] = v;
      } else if (col < 768) {                       // K
        const int cc = col - 384, hI = cc >> 6, d = cc & 63;
        Kb[(((size_t)bI * 6 + hI) * 256 + tok) * 64 + d] = v;
      } else {                                      // V, transposed store
        const int cc = col - 768, hI = cc >> 6, d = cc & 63;
        Vt[(((size_t)bI * 6 + hI) * 64 + d) * 256 + tok] = v;
      }
    }
  }
}

// ---------------------------------------------------------------------------
// Fused causal attention per (b,h). grid = B*H = 1536, block 256 (8 waves).
// Each wave owns q-blocks {wave, wave+8} of 16 queries.
// Computes S^T = K·Q^T so P fragments for O = P·V are in-register (no LDS).
// ---------------------------------------------------------------------------
__global__ void __launch_bounds__(256) k_attn(
    const _Float16* __restrict__ Qb, const _Float16* __restrict__ Kb,
    const _Float16* __restrict__ Vtb, _Float16* __restrict__ AO) {
  const int bh = blockIdx.x;
  const int bI = bh / 6, hI = bh % 6;
  const _Float16* Q = Qb + (size_t)bh * 256 * 64;
  const _Float16* K = Kb + (size_t)bh * 256 * 64;
  const _Float16* V = Vtb + (size_t)bh * 64 * 256;   // [64 x 256] (d-major)
  const int wave = threadIdx.x >> 5, l = threadIdx.x & 31;
  const int n = l & 15, sel = l >> 4;

  for (int qq = wave; qq < 16; qq += 8) {
    const int qbu = __builtin_amdgcn_readfirstlane(qq);  // uniform -> scalar branches
    const int q0 = qbu * 16;
    const int nT = qbu + 1;       // key tiles needed under causal mask
    f8v acc[16] = {};             // S^T tiles: acc[j] = keys [16j,16j+16) x q [q0,q0+16)

    // B fragments of Q^T reused across all key tiles
    h16 bq0 = frag_bt(Q, 64, q0, 0);
    h16 bq1 = frag_bt(Q, 64, q0, 32);
#pragma unroll
    for (int j = 0; j < 16; ++j) {
      if (j < nT) {                                   // uniform: EXEC stays full
        h16 a0 = frag_a(K, 64, j * 16, 0);
        acc[j] = wmma_f(a0, bq0, acc[j]);
        h16 a1 = frag_a(K, 64, j * 16, 32);
        acc[j] = wmma_f(a1, bq1, acc[j]);
      }
    }

    // Softmax over keys for this lane's query column q = q0 + n.
    // Lane holds keys {16j + 8*sel + r}; partner lane (l^16) holds the rest.
    const int qcol = q0 + n;
    float mx = -3.0e38f;
#pragma unroll
    for (int j = 0; j < 16; ++j)
      if (j < nT)
#pragma unroll
        for (int r = 0; r < 8; ++r) {
          const int key = j * 16 + sel * 8 + r;
          if (key <= qcol) mx = fmaxf(mx, acc[j][r]);
        }
    mx = fmaxf(mx, __shfl_xor(mx, 16, 32));
    float sum = 0.f;
#pragma unroll
    for (int j = 0; j < 16; ++j)
      if (j < nT)
#pragma unroll
        for (int r = 0; r < 8; ++r) {
          const int key = j * 16 + sel * 8 + r;
          const float p = (key <= qcol) ? __expf(acc[j][r] - mx) : 0.f;
          acc[j][r] = p;
          sum += p;
        }
    sum += __shfl_xor(sum, 16, 32);
    const float inv = 1.f / sum;
#pragma unroll
    for (int j = 0; j < 16; ++j)
      if (j < nT) acc[j] *= inv;
    // tiles j >= nT stay all-zero (fully masked) so PV can read them safely

    // O = P·V : A-frag of P comes straight from acc registers (D->A relayout
    // is the identity per-lane thanks to the S^T formulation).
    f8v o[4] = {};
    const int nS = (qbu >> 1) + 1;                    // 32-key steps needed
#pragma unroll
    for (int s = 0; s < 8; ++s) {
      if (s < nS) {
        h16 aP;
#pragma unroll
        for (int i = 0; i < 8; ++i) {
          aP[i] = (_Float16)acc[2 * s][i];
          aP[8 + i] = (_Float16)acc[2 * s + 1][i];
        }
#pragma unroll
        for (int d = 0; d < 4; ++d) {
          h16 bv = frag_bt(V, 256, d * 16, s * 32);   // Bt = V^T, contiguous in K
          o[d] = wmma_f(aP, bv, o[d]);
        }
      }
    }

    // Store to AO [B*T, 384] f16, row-major for the projection GEMM.
#pragma unroll
    for (int d = 0; d < 4; ++d)
#pragma unroll
      for (int r = 0; r < 8; ++r) {
        const int q = q0 + r + 8 * sel;
        AO[((size_t)bI * 256 + q) * 384 + hI * 64 + d * 16 + n] = (_Float16)o[d][r];
      }
  }
}

// ---------------------------------------------------------------------------
// GEMM3: out = AO[65536x384] @ Wproj + bproj (f32 out). grid (512, 6).
// ---------------------------------------------------------------------------
__global__ void __launch_bounds__(256) k_proj(
    const _Float16* __restrict__ AO, const _Float16* __restrict__ WT,
    const float* __restrict__ bias, float* __restrict__ out) {
  const int wave = threadIdx.x >> 5, l = threadIdx.x & 31;
  const int row0 = blockIdx.x * 128 + wave * 16;
  const int n0 = blockIdx.y * 64;
  f8v c[4] = {};
  for (int k0 = 0; k0 < 384; k0 += 32) {
    h16 a = frag_a(AO, 384, row0, k0);
#pragma unroll
    for (int t = 0; t < 4; ++t) {
      h16 b = frag_bt(WT, 384, n0 + t * 16, k0);
      c[t] = wmma_f(a, b, c[t]);
    }
  }
  const int n = l & 15, sel = l >> 4;
#pragma unroll
  for (int t = 0; t < 4; ++t) {
    const int col = n0 + t * 16 + n;
    const float bv = bias[col];
#pragma unroll
    for (int r = 0; r < 8; ++r) {
      const int row = row0 + r + 8 * sel;
      out[(size_t)row * 384 + col] = c[t][r] + bv;
    }
  }
}

// ---------------------------------------------------------------------------
extern "C" void kernel_launch(void* const* d_in, const int* in_sizes, int n_in,
                              void* d_out, int out_size, void* d_ws, size_t ws_size,
                              hipStream_t stream) {
  (void)in_sizes; (void)n_in; (void)out_size; (void)ws_size;
  const float* x     = (const float*)d_in[0];   // [256,256,384]
  const float* Wqkv  = (const float*)d_in[1];   // [384,1152]
  const float* Wproj = (const float*)d_in[2];   // [384,384]
  const float* bproj = (const float*)d_in[3];   // [384]
  float* out = (float*)d_out;

  const size_t NTOK = 256 * 256;                 // 65536 tokens
  char* ws = (char*)d_ws;
  size_t off = 0;
  auto carve = [&](size_t bytes) {
    void* p = ws + off;
    off += (bytes + 255) & ~(size_t)255;
    return p;
  };
  _Float16* xh  = (_Float16*)carve(NTOK * 384 * 2);          // x in f16
  _Float16* WqT = (_Float16*)carve((size_t)1152 * 384 * 2);  // Wqkv^T f16
  _Float16* WpT = (_Float16*)carve((size_t)384 * 384 * 2);   // Wproj^T f16
  _Float16* Qb  = (_Float16*)carve((size_t)256 * 6 * 256 * 64 * 2);
  _Float16* Kb  = (_Float16*)carve((size_t)256 * 6 * 256 * 64 * 2);
  _Float16* Vt  = (_Float16*)carve((size_t)256 * 6 * 64 * 256 * 2);
  _Float16* AO  = (_Float16*)carve(NTOK * 384 * 2);          // attention out f16

  const int nx = (int)(NTOK * 384);
  k_f32_to_f16<<<(nx + 255) / 256, 256, 0, stream>>>(x, xh, nx);
  k_transpose_f16<<<(384 * 1152 + 255) / 256, 256, 0, stream>>>(Wqkv, WqT, 384, 1152);
  k_transpose_f16<<<(384 * 384 + 255) / 256, 256, 0, stream>>>(Wproj, WpT, 384, 384);

  k_gemm_qkv<<<dim3(512, 18), 256, 0, stream>>>(xh, WqT, Qb, Kb, Vt);
  k_attn<<<dim3(1536), 256, 0, stream>>>(Qb, Kb, Vt, AO);
  k_proj<<<dim3(512, 6), 256, 0, stream>>>(AO, WpT, bproj, out);
}
